// MaskedGRUCell_31035433681037
// MI455X (gfx1250) — matrix operand order repaired and checked by reading, not compiled, not run on hardware
//
#include <hip/hip_runtime.h>

// ---------------------------------------------------------------------------
// MaskedGRUCell, B=131072, H=256.
// Diagonal-masked weights => elementwise; only Wmu/Wg are dense GEMMs.
// GEMMs via v_wmma_f32_16x16x32_bf16 with hi/lo bf16 split (fp32-accurate):
//   x*W ~= xh*Wh + xh*Wl + xl*Wh   (error O(2^-18))
// Fully fused: GEMM + GRU + masked exp decay in one kernel.
// ---------------------------------------------------------------------------

#define HDIM 256
#define MTILE 32   // batch rows per block
#define NPW   32   // output columns per wave (2 wmma N-tiles)

typedef __attribute__((ext_vector_type(16))) __bf16 bf16x16;
typedef __attribute__((ext_vector_type(8)))  __bf16 bf16x8;
typedef __attribute__((ext_vector_type(8)))  float  f32x8;

static __device__ __forceinline__ __bf16 f2bf(float f) {
    union { float f; unsigned u; } a; a.f = f;
    unsigned r = (a.u + 0x7FFFu + ((a.u >> 16) & 1u)) >> 16;   // RNE
    union { unsigned short s; __bf16 b; } o; o.s = (unsigned short)r;
    return o.b;
}
static __device__ __forceinline__ float bf2f(__bf16 b) {
    union { unsigned short s; __bf16 b; } i; i.b = b;
    union { unsigned u; float f; } o; o.u = ((unsigned)i.s) << 16;
    return o.f;
}
static __device__ __forceinline__ f32x8 wmma_bf16(bf16x16 a, bf16x16 b, f32x8 c) {
    return __builtin_amdgcn_wmma_f32_16x16x32_bf16(
        /*neg_a=*/false, a, /*neg_b=*/false, b,
        /*c_mod=*/(short)0, c, /*reuse_a=*/false, /*reuse_b=*/false);
}
static __device__ __forceinline__ float sigmoidf_(float v) {
    return 1.0f / (1.0f + __expf(-v));
}

// A fragment (16-bit 16x32 layout): lane half h, row m = lane&15.
// elements 0..7  <- K = k0 + 8h + e        (16B contiguous)
// elements 8..15 <- K = k0 + 16 + 8h + e-8 (16B contiguous)
static __device__ __forceinline__ bf16x16 load_a_frag(const __bf16* s, int base0) {
    bf16x8 lo = *(const bf16x8*)(s + base0);
    bf16x8 hi = *(const bf16x8*)(s + base0 + 16);
    bf16x16 r;
#pragma unroll
    for (int e = 0; e < 8; ++e) { r[e] = lo[e]; r[e + 8] = hi[e]; }
    return r;
}

// B fragment (32x16): lane half h, col n = lane&15.
// elements 0..7 <- K = k0 + 8h + e, 8..15 <- K = k0 + 16 + 8h + e-8 (W[n][k] row-major)
static __device__ __forceinline__ bf16x16 load_b_frag(const __bf16* w, int n, int kbase) {
    const __bf16* p = w + n * HDIM + kbase;
    bf16x8 lo = *(const bf16x8*)(p);
    bf16x8 hi = *(const bf16x8*)(p + 16);
    bf16x16 r;
#pragma unroll
    for (int e = 0; e < 8; ++e) { r[e] = lo[e]; r[e + 8] = hi[e]; }
    return r;
}

// ---------------------------------------------------------------------------
// Prologue: bf16 hi/lo weight copies + contiguous diagonal table in d_ws.
// ---------------------------------------------------------------------------
__global__ __launch_bounds__(256) void maskedgru_prep(
    const float* __restrict__ Wmu, const float* __restrict__ Wg,
    const float* __restrict__ Wir, const float* __restrict__ Whr,
    const float* __restrict__ Wiz, const float* __restrict__ Whz,
    const float* __restrict__ Win, const float* __restrict__ Whn,
    __bf16* __restrict__ mu_hi, __bf16* __restrict__ mu_lo,
    __bf16* __restrict__ g_hi,  __bf16* __restrict__ g_lo,
    float* __restrict__ diag)
{
    int i = blockIdx.x * 256 + threadIdx.x;     // 0 .. H*H-1
    float wm = Wmu[i];
    __bf16 mh = f2bf(wm);
    mu_hi[i] = mh;
    mu_lo[i] = f2bf(wm - bf2f(mh));
    float wg = Wg[i];
    __bf16 gh = f2bf(wg);
    g_hi[i] = gh;
    g_lo[i] = f2bf(wg - bf2f(gh));
    if (i < HDIM) {
        int d = i * HDIM + i;
        diag[0 * HDIM + i] = Wir[d];
        diag[1 * HDIM + i] = Whr[d];
        diag[2 * HDIM + i] = Wiz[d];
        diag[3 * HDIM + i] = Whz[d];
        diag[4 * HDIM + i] = Win[d];
        diag[5 * HDIM + i] = Whn[d];
    }
}

// ---------------------------------------------------------------------------
// Fused main kernel: 256 threads (8 wave32), 32 batch rows per block.
// ---------------------------------------------------------------------------
__global__ __launch_bounds__(256) void maskedgru_fused(
    const float* __restrict__ x_t, const float* __restrict__ h_prev,
    const float* __restrict__ delta_t, const float* __restrict__ m_t,
    const float* __restrict__ Wir_b, const float* __restrict__ Wiz_b,
    const float* __restrict__ Win_b,
    const float* __restrict__ Wmu_b, const float* __restrict__ Wg_b,
    const __bf16* __restrict__ mu_hi, const __bf16* __restrict__ mu_lo,
    const __bf16* __restrict__ g_hi,  const __bf16* __restrict__ g_lo,
    const float* __restrict__ diag,
    float* __restrict__ out)
{
    __shared__ __bf16 sxh[MTILE * HDIM];
    __shared__ __bf16 sxl[MTILE * HDIM];
    __shared__ __bf16 sdh[MTILE * HDIM];
    __shared__ __bf16 sdl[MTILE * HDIM];

    const int tid = threadIdx.x;
    const size_t rowBase = (size_t)blockIdx.x * MTILE;

    // Stage x / delta tiles as bf16 hi/lo in LDS (shared by all 8 waves).
#pragma unroll 4
    for (int i = tid; i < MTILE * HDIM; i += 256) {
        size_t g = rowBase * HDIM + (size_t)i;
        float xv = x_t[g];
        __bf16 xh = f2bf(xv);
        sxh[i] = xh;
        sxl[i] = f2bf(xv - bf2f(xh));
        float dv = delta_t[g];
        __bf16 dh = f2bf(dv);
        sdh[i] = dh;
        sdl[i] = f2bf(dv - bf2f(dh));
    }
    __syncthreads();

    const int wave = tid >> 5;
    const int lane = tid & 31;
    const int half = lane >> 4;
    const int l16  = lane & 15;
    const int n0   = wave * NPW;          // this wave's 32 output columns

    f32x8 accm[2][2] = {};                // [mtile][ntile] mu accumulators
    f32x8 accg[2][2] = {};                // [mtile][ntile] gamma accumulators

    for (int k0 = 0; k0 < HDIM; k0 += 32) {
        const int abase = k0 + half * 8;
        const int kb    = k0 + half * 8;

        bf16x16 axh[2], axl[2], adh[2], adl[2];
#pragma unroll
        for (int mt = 0; mt < 2; ++mt) {
            int m = mt * 16 + l16;
            axh[mt] = load_a_frag(sxh, m * HDIM + abase);
            axl[mt] = load_a_frag(sxl, m * HDIM + abase);
            adh[mt] = load_a_frag(sdh, m * HDIM + abase);
            adl[mt] = load_a_frag(sdl, m * HDIM + abase);
        }

#pragma unroll
        for (int t = 0; t < 2; ++t) {
            int n = n0 + t * 16 + l16;
            bf16x16 bmh = load_b_frag(mu_hi, n, kb);
            bf16x16 bml = load_b_frag(mu_lo, n, kb);
#pragma unroll
            for (int mt = 0; mt < 2; ++mt) {
                accm[mt][t] = wmma_bf16(axh[mt], bmh, accm[mt][t]);
                accm[mt][t] = wmma_bf16(axh[mt], bml, accm[mt][t]);
                accm[mt][t] = wmma_bf16(axl[mt], bmh, accm[mt][t]);
            }
            bf16x16 bgh = load_b_frag(g_hi, n, kb);
            bf16x16 bgl = load_b_frag(g_lo, n, kb);
#pragma unroll
            for (int mt = 0; mt < 2; ++mt) {
                accg[mt][t] = wmma_bf16(adh[mt], bgh, accg[mt][t]);
                accg[mt][t] = wmma_bf16(adh[mt], bgl, accg[mt][t]);
                accg[mt][t] = wmma_bf16(adl[mt], bgh, accg[mt][t]);
            }
        }
    }

    // Epilogue: D layout -> lane l16 = column, VGPR v -> row v + 8*half.
#pragma unroll
    for (int t = 0; t < 2; ++t) {
        const int col = n0 + t * 16 + l16;
        const float dir = diag[0 * HDIM + col];
        const float dhr = diag[1 * HDIM + col];
        const float diz = diag[2 * HDIM + col];
        const float dhz = diag[3 * HDIM + col];
        const float din = diag[4 * HDIM + col];
        const float dhn = diag[5 * HDIM + col];
        const float bir = Wir_b[col];
        const float biz = Wiz_b[col];
        const float bin = Win_b[col];
        const float bmu = Wmu_b[col];
        const float bg  = Wg_b[col];
#pragma unroll
        for (int mt = 0; mt < 2; ++mt) {
#pragma unroll
            for (int v = 0; v < 8; ++v) {
                const int rl = mt * 16 + v + half * 8;
                const size_t g = (rowBase + (size_t)rl) * HDIM + (size_t)col;
                const float xv = x_t[g];
                const float hv = h_prev[g];
                const float mv = m_t[g];
                const float mu = accm[mt][t][v] + bmu;
                const float ga = accg[mt][t][v] + bg;
                const float r  = sigmoidf_(xv * dir + bir + hv * dhr);
                const float z  = sigmoidf_(xv * diz + biz + hv * dhz);
                const float nn = tanhf(xv * din + bin + hv * r * dhn);
                const float ht = (1.0f - z) * hv + z * nn;
                out[g] = mu + (ht - mu) * __expf(-ga * mv);
            }
        }
    }
}

// ---------------------------------------------------------------------------
extern "C" void kernel_launch(void* const* d_in, const int* in_sizes, int n_in,
                              void* d_out, int out_size, void* d_ws, size_t ws_size,
                              hipStream_t stream) {
    const float* x_t     = (const float*)d_in[0];
    const float* h_prev  = (const float*)d_in[1];
    const float* delta_t = (const float*)d_in[2];
    const float* m_t     = (const float*)d_in[3];
    const float* Wir_w   = (const float*)d_in[4];
    const float* Wir_b   = (const float*)d_in[5];
    const float* Wiz_w   = (const float*)d_in[6];
    const float* Wiz_b   = (const float*)d_in[7];
    const float* Win_w   = (const float*)d_in[8];
    const float* Win_b   = (const float*)d_in[9];
    const float* Whr_w   = (const float*)d_in[10];
    const float* Whz_w   = (const float*)d_in[11];
    const float* Whn_w   = (const float*)d_in[12];
    const float* Wmu_w   = (const float*)d_in[13];
    const float* Wmu_b   = (const float*)d_in[14];
    const float* Wg_w    = (const float*)d_in[15];
    const float* Wg_b    = (const float*)d_in[16];

    char* ws = (char*)d_ws;
    __bf16* mu_hi = (__bf16*)(ws + 0 * 128 * 1024);
    __bf16* mu_lo = (__bf16*)(ws + 1 * 128 * 1024);
    __bf16* g_hi  = (__bf16*)(ws + 2 * 128 * 1024);
    __bf16* g_lo  = (__bf16*)(ws + 3 * 128 * 1024);
    float*  diag  = (float*)(ws + 4 * 128 * 1024);

    const int B = in_sizes[0] / HDIM;   // 131072

    maskedgru_prep<<<(HDIM * HDIM) / 256, 256, 0, stream>>>(
        Wmu_w, Wg_w, Wir_w, Whr_w, Wiz_w, Whz_w, Win_w, Whn_w,
        mu_hi, mu_lo, g_hi, g_lo, diag);

    maskedgru_fused<<<B / MTILE, 256, 0, stream>>>(
        x_t, h_prev, delta_t, m_t,
        Wir_b, Wiz_b, Win_b, Wmu_b, Wg_b,
        mu_hi, mu_lo, g_hi, g_lo, diag,
        (float*)d_out);
}
